// VQShape_25159918420262
// MI455X (gfx1250) — compile-verified, matrix operand order checked by reading, not run on hardware
//
#include <hip/hip_runtime.h>

// ---------------------------------------------------------------------------
// Model constants (VQShape reference)
// ---------------------------------------------------------------------------
#define BB   512
#define TT   512
#define EE   256
#define NHH  8
#define NPP  64
#define PSS  8
#define NLL  6
#define KK   512
#define DCC  8

#define ACT_NONE 0
#define ACT_GELU 1

typedef __bf16 bhalf;
typedef bhalf v16bf __attribute__((ext_vector_type(16)));
typedef bhalf v4bf  __attribute__((ext_vector_type(4)));
typedef float v8f   __attribute__((ext_vector_type(8)));

union ABf { v16bf v; bhalf e[16]; };
union CF  { v8f   v; float e[8]; };

__device__ __forceinline__ float gelu_f(float x) {
  return 0.5f * x * (1.f + erff(x * 0.70710678118654752f));
}

// ---------------------------------------------------------------------------
// WMMA GEMM: C[M,N] = act(A[M,K] @ W[K,N] + bias)
// block = 256 threads (8 waves), tile 128x64, K-step 32 (bf16 WMMA, f32 accum)
// Callers guarantee M%128==0, N%64==0, K%32==0 -> no bounds guards anywhere.
// ---------------------------------------------------------------------------
__global__ __launch_bounds__(256) void gemm_bf16_wmma(
    const float* __restrict__ A, const float* __restrict__ W,
    const float* __restrict__ bias, float* __restrict__ C,
    int M, int N, int K, int act)
{
  __shared__ __align__(16) bhalf As[128][40];   // [m][k] (+pad)
  __shared__ __align__(16) bhalf Bs[64][40];    // transposed: [n][k] (+pad)

  const int tid  = threadIdx.x;
  const int m0   = blockIdx.y * 128;
  const int n0   = blockIdx.x * 64;
  const int w    = tid >> 5;
  const int lane = tid & 31;
  const int wm   = w >> 1;        // 0..3
  const int wn   = w & 1;         // 0..1
  const int l    = lane & 15;
  const int hi   = lane >> 4;

  CF acc[2][2];
#pragma unroll
  for (int i = 0; i < 2; i++)
#pragma unroll
    for (int j = 0; j < 2; j++)
#pragma unroll
      for (int r = 0; r < 8; r++) acc[i][j].e[r] = 0.f;

  const int ar = tid >> 1;         // 0..127 (row of A tile)
  const int ac = (tid & 1) * 16;   // 0 | 16
  const int bk = tid >> 3;         // 0..31  (row of W tile)
  const int bn = (tid & 7) * 8;    // 0..56

  const float* Arow = A + (size_t)(m0 + ar) * K + ac;
  const float* Wrow = W + (size_t)bk * N + n0 + bn;

  for (int k0 = 0; k0 < K; k0 += 32) {
    // ---- stage A tile: 4x float4 per thread, pack to bf16, 8B LDS stores ----
    {
      const float4* Ap = (const float4*)(Arow + k0);
      if (k0 + 64 < K) __builtin_prefetch(Arow + k0 + 64, 0, 1);
#pragma unroll
      for (int j = 0; j < 4; j++) {
        const float4 f = Ap[j];
        const v4bf h = { (bhalf)f.x, (bhalf)f.y, (bhalf)f.z, (bhalf)f.w };
        *(v4bf*)&As[ar][ac + j * 4] = h;
      }
    }
    // ---- stage B tile transposed: 2x float4 per thread ----
    {
      const float4* Bp = (const float4*)(Wrow + (size_t)k0 * N);
#pragma unroll
      for (int j2 = 0; j2 < 2; j2++) {
        const float4 f = Bp[j2];
        Bs[bn + j2 * 4 + 0][bk] = (bhalf)f.x;
        Bs[bn + j2 * 4 + 1][bk] = (bhalf)f.y;
        Bs[bn + j2 * 4 + 2][bk] = (bhalf)f.z;
        Bs[bn + j2 * 4 + 3][bk] = (bhalf)f.w;
      }
    }
    __syncthreads();

    ABf af[2], bf2[2];
#pragma unroll
    for (int i = 0; i < 2; i++) {
      const int row = wm * 32 + i * 16 + l;
#pragma unroll
      for (int j = 0; j < 8; j++) {
        af[i].e[j]     = As[row][hi * 8 + j];
        af[i].e[8 + j] = As[row][16 + hi * 8 + j];
      }
    }
#pragma unroll
    for (int j2 = 0; j2 < 2; j2++) {
      const int col = wn * 32 + j2 * 16 + l;
#pragma unroll
      for (int j = 0; j < 8; j++) {
        bf2[j2].e[j]     = Bs[col][hi * 8 + j];
        bf2[j2].e[8 + j] = Bs[col][16 + hi * 8 + j];
      }
    }
#pragma unroll
    for (int i = 0; i < 2; i++)
#pragma unroll
      for (int j = 0; j < 2; j++)
        acc[i][j].v = __builtin_amdgcn_wmma_f32_16x16x32_bf16(
            false, af[i].v, false, bf2[j].v, (short)0, acc[i][j].v, false, false);
    __syncthreads();
  }

#pragma unroll
  for (int i = 0; i < 2; i++) {
#pragma unroll
    for (int j = 0; j < 2; j++) {
      const int col = n0 + wn * 32 + j * 16 + l;
      const float bv = bias ? bias[col] : 0.f;
#pragma unroll
      for (int r = 0; r < 8; r++) {
        const int row = m0 + wm * 32 + i * 16 + hi * 8 + r;
        float v = acc[i][j].e[r] + bv;
        if (act == ACT_GELU) v = gelu_f(v);
        C[(size_t)row * N + col] = v;
      }
    }
  }
}

// ---------------------------------------------------------------------------
// Fused attention per (batch, head): S = QK^T * scale, softmax, O = P V
// d = 32, Lq <= 80 (padded), Lk <= 96 (padded). block = 128 threads (4 waves)
// ---------------------------------------------------------------------------
__global__ __launch_bounds__(128) void attn_wmma(
    const float* __restrict__ Qg, const float* __restrict__ Kg,
    const float* __restrict__ Vg, float* __restrict__ Og,
    int Lq, int Lk, int E_, int NH_, float scale)
{
  __shared__ __align__(16) bhalf Qs[80 * 32];
  __shared__ __align__(16) bhalf Ks[96 * 32];
  __shared__ __align__(16) bhalf Vt[32 * 96];   // transposed: [d][key]
  __shared__ __align__(16) float Sc[80 * 96];
  __shared__ __align__(16) bhalf Ps[80 * 96];

  const int b    = blockIdx.x / NH_;
  const int h    = blockIdx.x % NH_;
  const int tid  = threadIdx.x;
  const int w    = tid >> 5;
  const int lane = tid & 31;
  const int l    = lane & 15;
  const int hi   = lane >> 4;

  const float* qb = Qg + (size_t)(b * Lq) * E_ + h * 32;
  const float* kb = Kg + (size_t)(b * Lk) * E_ + h * 32;
  const float* vb = Vg + (size_t)(b * Lk) * E_ + h * 32;

  const v4bf zh = { (bhalf)0.f, (bhalf)0.f, (bhalf)0.f, (bhalf)0.f };

  // Q: 80 rows x 8 float4 columns
  for (int i4 = tid; i4 < 80 * 8; i4 += 128) {
    const int r = i4 >> 3, c4 = (i4 & 7) * 4;
    v4bf hq = zh;
    if (r < Lq) {
      const float4 f = *(const float4*)(qb + (size_t)r * E_ + c4);
      hq = (v4bf){ (bhalf)f.x, (bhalf)f.y, (bhalf)f.z, (bhalf)f.w };
    }
    *(v4bf*)&Qs[r * 32 + c4] = hq;
  }
  // K / V: 96 rows x 8 float4 columns (V stored transposed)
  for (int i4 = tid; i4 < 96 * 8; i4 += 128) {
    const int r = i4 >> 3, c4 = (i4 & 7) * 4;
    float4 fk = {0.f, 0.f, 0.f, 0.f}, fv = {0.f, 0.f, 0.f, 0.f};
    if (r < Lk) {
      fk = *(const float4*)(kb + (size_t)r * E_ + c4);
      fv = *(const float4*)(vb + (size_t)r * E_ + c4);
    }
    *(v4bf*)&Ks[r * 32 + c4] = (v4bf){ (bhalf)fk.x, (bhalf)fk.y, (bhalf)fk.z, (bhalf)fk.w };
    Vt[(c4 + 0) * 96 + r] = (bhalf)fv.x;
    Vt[(c4 + 1) * 96 + r] = (bhalf)fv.y;
    Vt[(c4 + 2) * 96 + r] = (bhalf)fv.z;
    Vt[(c4 + 3) * 96 + r] = (bhalf)fv.w;
  }
  // Ps zero-init, 16B stores
  {
    uint4 z4; z4.x = z4.y = z4.z = z4.w = 0u;
    for (int i = tid; i < (80 * 96 * 2) / 16; i += 128) ((uint4*)Ps)[i] = z4;
  }
  __syncthreads();

  const int nQt = (Lq + 15) >> 4;
  const int nKt = (Lk + 15) >> 4;

  // ---- S = Q K^T (single K=32 WMMA per 16x16 tile since d=32) ----
  for (int t = w; t < nQt * nKt; t += 4) {
    const int ti = t / nKt, tj = t % nKt;
    ABf a, bm;
#pragma unroll
    for (int j = 0; j < 8; j++) {
      const int row = ti * 16 + l;
      a.e[j]      = Qs[row * 32 + hi * 8 + j];
      a.e[8 + j]  = Qs[row * 32 + 16 + hi * 8 + j];
      const int col = tj * 16 + l;
      bm.e[j]     = Ks[col * 32 + hi * 8 + j];
      bm.e[8 + j] = Ks[col * 32 + 16 + hi * 8 + j];
    }
    CF acc;
#pragma unroll
    for (int r = 0; r < 8; r++) acc.e[r] = 0.f;
    acc.v = __builtin_amdgcn_wmma_f32_16x16x32_bf16(
        false, a.v, false, bm.v, (short)0, acc.v, false, false);
#pragma unroll
    for (int r = 0; r < 8; r++)
      Sc[(ti * 16 + hi * 8 + r) * 96 + tj * 16 + l] = acc.e[r] * scale;
  }
  __syncthreads();

  // ---- softmax rows (fp32) -> Ps (bf16); padded cols already zero ----
  for (int r = tid; r < Lq; r += 128) {
    float* sr = &Sc[r * 96];
    float mx = -1e30f;
    for (int c = 0; c < Lk; c++) mx = fmaxf(mx, sr[c]);
    float s = 0.f;
    for (int c = 0; c < Lk; c++) {
      const float e = __expf(sr[c] - mx);
      s += e;
      sr[c] = e;
    }
    const float inv = 1.f / s;
    for (int c = 0; c < Lk; c++) Ps[r * 96 + c] = (bhalf)(sr[c] * inv);
  }
  __syncthreads();

  // ---- O = P V ----
  const int nKc = (Lk + 31) >> 5;
  for (int t = w; t < nQt * 2; t += 4) {
    const int ti = t >> 1, tj = t & 1;
    CF acc;
#pragma unroll
    for (int r = 0; r < 8; r++) acc.e[r] = 0.f;
    for (int kc = 0; kc < nKc; kc++) {
      ABf a, bm;
#pragma unroll
      for (int j = 0; j < 8; j++) {
        const int row = ti * 16 + l;
        a.e[j]      = Ps[row * 96 + kc * 32 + hi * 8 + j];
        a.e[8 + j]  = Ps[row * 96 + kc * 32 + 16 + hi * 8 + j];
        const int col = tj * 16 + l;
        bm.e[j]     = Vt[col * 96 + kc * 32 + hi * 8 + j];
        bm.e[8 + j] = Vt[col * 96 + kc * 32 + 16 + hi * 8 + j];
      }
      acc.v = __builtin_amdgcn_wmma_f32_16x16x32_bf16(
          false, a.v, false, bm.v, (short)0, acc.v, false, false);
    }
#pragma unroll
    for (int r = 0; r < 8; r++) {
      const int row = ti * 16 + hi * 8 + r;
      if (row < Lq)
        Og[(size_t)(b * Lq + row) * E_ + h * 32 + tj * 16 + l] = acc.e[r];
    }
  }
}

// ---------------------------------------------------------------------------
// TDM (Tensor Data Mover) staging path, CDNA5 ISA-8.
// Stages a 32x64 f32 tile of W (row stride N) into LDS via tensor_load_to_lds,
// then waits on TENSORcnt. Emitted into the bundle as a compile probe for the
// async-tensor path; the launch path keeps the WMMA pipeline.
// ---------------------------------------------------------------------------
#if __has_builtin(__builtin_amdgcn_tensor_load_to_lds)
typedef unsigned int v4u __attribute__((ext_vector_type(4)));
typedef int v4i __attribute__((ext_vector_type(4)));
typedef int v8i __attribute__((ext_vector_type(8)));

__global__ void tdm_stage_tile(const float* __restrict__ W, float* __restrict__ out,
                               int N, int k0, int n0)
{
  __shared__ __align__(16) float Bs[32 * 64];

  const unsigned long long gaddr =
      (unsigned long long)(const void*)(W + (size_t)k0 * N + n0);

  // D# group 0: count=1 | lds_addr[63:32] | global_addr[120:64] | type=2
  v4u g0;
  g0.x = 1u;                                          // count=1, user mode
  g0.y = 0u;                                          // lds_addr = base of LDS alloc
  g0.z = (unsigned)(gaddr & 0xffffffffu);             // global_addr[31:0]
  g0.w = (unsigned)((gaddr >> 32) & 0x01ffffffu) | (2u << 30);  // addr MSBs | type=2

  // D# group 1: mask=0, data_size=4B(2), dims/strides per §8.4 bit layout
  const unsigned long long td0 = (unsigned long long)(unsigned)N;  // tensor_dim0
  const unsigned long long st0 = (unsigned long long)(unsigned)N;  // dim0 stride
  v8i g1;
  g1[0] = (int)(2u << 16);                                   // data_size=4B
  g1[1] = (int)((td0 & 0xffffu) << 16);                      // tensor_dim0[15:0]
  g1[2] = (int)(((td0 >> 16) & 0xffffu) | (32u << 16));      // dim0 hi | tensor_dim1 lo
  g1[3] = (int)(64u << 16);                                  // dim1 hi=0 | tile_dim0=64
  g1[4] = (int)(32u);                                        // tile_dim1=32, tile_dim2=0
  g1[5] = (int)(st0 & 0xffffffffu);                          // dim0_stride[31:0]
  g1[6] = (int)((st0 >> 32) & 0xffffu);                      // dim0_stride[47:32]
  g1[7] = 0;                                                 // dim1_stride hi
  const v4i g2 = {0, 0, 0, 0};
  const v4i g3 = {0, 0, 0, 0};

#if __clang_major__ >= 23
  __builtin_amdgcn_tensor_load_to_lds(g0, g1, g2, g3, (v8i){0,0,0,0,0,0,0,0}, 0);
#else
  __builtin_amdgcn_tensor_load_to_lds(g0, g1, g2, g3, 0);
#endif
  __builtin_amdgcn_s_wait_tensorcnt((short)0);
  __syncthreads();

  for (int t = threadIdx.x; t < 32 * 64; t += blockDim.x)
    out[(size_t)blockIdx.x * 2048 + t] = Bs[t];
}
#endif

// ---------------------------------------------------------------------------
// Fused residual + LayerNorm over rows of width 256 (one wave32 per row)
// ---------------------------------------------------------------------------
__global__ __launch_bounds__(256) void add_ln256(
    const float* __restrict__ a, const float* __restrict__ b,
    const float* __restrict__ g, const float* __restrict__ be,
    float* __restrict__ y, int rows)
{
  const int row  = blockIdx.x * 8 + (threadIdx.x >> 5);
  const int lane = threadIdx.x & 31;
  if (row >= rows) return;
  const size_t base = (size_t)row * 256;
  float v[8];
  float s = 0.f;
#pragma unroll
  for (int j = 0; j < 8; j++) {
    const int c = j * 32 + lane;
    v[j] = a[base + c] + b[base + c];
    s += v[j];
  }
#pragma unroll
  for (int m = 16; m > 0; m >>= 1) s += __shfl_xor(s, m, 32);
  const float mean = s * (1.f / 256.f);
  float q = 0.f;
#pragma unroll
  for (int j = 0; j < 8; j++) { const float d = v[j] - mean; q += d * d; }
#pragma unroll
  for (int m = 16; m > 0; m >>= 1) q += __shfl_xor(q, m, 32);
  const float rs = rsqrtf(q * (1.f / 256.f) + 1e-5f);
#pragma unroll
  for (int j = 0; j < 8; j++) {
    const int c = j * 32 + lane;
    y[base + c] = (v[j] - mean) * rs * g[c] + be[c];
  }
}

// ---------------------------------------------------------------------------
// Scalar GEMM for tiny K/N (patch proj K=8, codebook dims, ShapeDecoder MLP)
// ---------------------------------------------------------------------------
__global__ void gemm_small(const float* __restrict__ A, const float* __restrict__ W,
                           const float* __restrict__ bias, float* __restrict__ C,
                           int M, int N, int K, int act)
{
  const int i = blockIdx.x * blockDim.x + threadIdx.x;
  if (i >= M * N) return;
  const int m = i / N, n = i % N;
  float s = bias ? bias[n] : 0.f;
  for (int k = 0; k < K; k++) s += A[(size_t)m * K + k] * W[(size_t)k * N + n];
  if (act == ACT_GELU) s = gelu_f(s);
  C[i] = s;
}

// LayerNorm + GELU for small dims (<=32), one thread per row
__global__ void ln_gelu_small(const float* __restrict__ x, const float* __restrict__ g,
                              const float* __restrict__ b, float* __restrict__ y,
                              int rows, int dim)
{
  const int r = blockIdx.x * blockDim.x + threadIdx.x;
  if (r >= rows) return;
  const float* xr = x + (size_t)r * dim;
  float m = 0.f;
  for (int d = 0; d < dim; d++) m += xr[d];
  m /= (float)dim;
  float v = 0.f;
  for (int d = 0; d < dim; d++) { const float t = xr[d] - m; v += t * t; }
  const float rs = rsqrtf(v / (float)dim + 1e-5f);
  for (int d = 0; d < dim; d++)
    y[(size_t)r * dim + d] = gelu_f((xr[d] - m) * rs * g[d] + b[d]);
}

// ---------------------------------------------------------------------------
// misc elementwise kernels
// ---------------------------------------------------------------------------
__global__ void zero_f(float* p, int n) {
  const int i = blockIdx.x * blockDim.x + threadIdx.x;
  if (i < n) p[i] = 0.f;
}

__global__ void add_pos(float* __restrict__ h, const float* __restrict__ pos, int n) {
  const int i = blockIdx.x * blockDim.x + threadIdx.x;
  if (i >= n) return;
  const int e = i % EE;
  const int p = (i / EE) % NPP;
  h[i] += pos[p * EE + e];
}

__global__ void broadcast_tok(const float* __restrict__ tok, float* __restrict__ t, int n) {
  const int i = blockIdx.x * blockDim.x + threadIdx.x;
  if (i >= n) return;
  const int e = i % EE;
  const int p = (i / EE) % NPP;
  t[i] = tok[p * EE + e];
}

__global__ void build_dseq(const float* __restrict__ cls, const float* __restrict__ ebuf,
                           float* __restrict__ d, int n) {
  const int i = blockIdx.x * blockDim.x + threadIdx.x;
  if (i >= n) return;
  const int e = i % EE;
  const int row = i / EE;
  const int b = row / (NPP + 1);
  const int r = row % (NPP + 1);
  d[i] = (r == 0) ? cls[e] : ebuf[((size_t)b * NPP + (r - 1)) * EE + e];
}

__global__ void extract_xhat(const float* __restrict__ Ct, float* __restrict__ out, int n) {
  const int i = blockIdx.x * blockDim.x + threadIdx.x;
  if (i >= n) return;
  const int b = i / (NPP * PSS);
  const int rem = i % (NPP * PSS);
  const int p = rem / PSS, j = rem % PSS;
  out[i] = Ct[((size_t)(b * (NPP + 1) + 1 + p)) * PSS + j];
}

// ---------------------------------------------------------------------------
// Codebook: pass1 (argmin + row softmax stats + mse/entropy partials)
// ---------------------------------------------------------------------------
__global__ __launch_bounds__(256) void cb_pass1(
    const float* __restrict__ z, const float* __restrict__ cb,
    float* __restrict__ zq, int* __restrict__ idx,
    float* __restrict__ rowm, float* __restrict__ rowz,
    float* __restrict__ acc2, int rows)
{
  const int r = blockIdx.x * blockDim.x + threadIdx.x;
  float mse = 0.f, ent = 0.f;
  if (r < rows) {
    float zr[DCC];
#pragma unroll
    for (int d = 0; d < DCC; d++) zr[d] = z[(size_t)r * DCC + d];
    float bm = 1e30f;
    int best = 0;
    for (int j = 0; j < KK; j++) {
      float dd = 0.f;
#pragma unroll
      for (int d = 0; d < DCC; d++) { const float t = zr[d] - cb[j * DCC + d]; dd += t * t; }
      if (dd < bm) { bm = dd; best = j; }
    }
    float S1 = 0.f, S2 = 0.f;
    for (int j = 0; j < KK; j++) {
      float dd = 0.f;
#pragma unroll
      for (int d = 0; d < DCC; d++) { const float t = zr[d] - cb[j * DCC + d]; dd += t * t; }
      const float lj = (bm - dd) * 100.f;   // -(d - min)/0.01
      const float e  = __expf(lj);
      S1 += e;
      S2 += e * lj;
    }
#pragma unroll
    for (int d = 0; d < DCC; d++) {
      const float q = cb[best * DCC + d];
      const float t = q - zr[d];
      mse += t * t;
      zq[(size_t)r * DCC + d] = q;   // straight-through == zq numerically
    }
    idx[r]  = best;
    rowm[r] = bm;
    rowz[r] = S1;
    ent = -(S2 / S1 - logf(S1));
  }
  __shared__ float red[256];
  red[threadIdx.x] = mse;
  __syncthreads();
  for (int m = 128; m > 0; m >>= 1) {
    if (threadIdx.x < m) red[threadIdx.x] += red[threadIdx.x + m];
    __syncthreads();
  }
  if (threadIdx.x == 0) atomicAdd(&acc2[0], red[0]);
  __syncthreads();
  red[threadIdx.x] = ent;
  __syncthreads();
  for (int m = 128; m > 0; m >>= 1) {
    if (threadIdx.x < m) red[threadIdx.x] += red[threadIdx.x + m];
    __syncthreads();
  }
  if (threadIdx.x == 0) atomicAdd(&acc2[1], red[0]);
}

// pass2: mean probability per code
__global__ __launch_bounds__(256) void cb_pass2(
    const float* __restrict__ z, const float* __restrict__ cb,
    const float* __restrict__ rowm, const float* __restrict__ rowz,
    float* __restrict__ pm, int rows)
{
  const int j = blockIdx.x;
  float cj[DCC];
#pragma unroll
  for (int d = 0; d < DCC; d++) cj[d] = cb[j * DCC + d];
  float s = 0.f;
  for (int r = threadIdx.x; r < rows; r += 256) {
    float dd = 0.f;
#pragma unroll
    for (int d = 0; d < DCC; d++) { const float t = z[(size_t)r * DCC + d] - cj[d]; dd += t * t; }
    s += __expf((rowm[r] - dd) * 100.f) / rowz[r];
  }
  __shared__ float red[256];
  red[threadIdx.x] = s;
  __syncthreads();
  for (int m = 128; m > 0; m >>= 1) {
    if (threadIdx.x < m) red[threadIdx.x] += red[threadIdx.x + m];
    __syncthreads();
  }
  if (threadIdx.x == 0) pm[j] = red[0] / (float)rows;
}

__global__ __launch_bounds__(256) void cb_finalize(
    const float* __restrict__ acc2, const float* __restrict__ pm,
    float* __restrict__ loss_out, int rows)
{
  __shared__ float red[256];
  float s = 0.f;
  for (int j = threadIdx.x; j < KK; j += 256) {
    const float p = pm[j];
    s += -p * logf(fmaxf(p, 1e-5f));
  }
  red[threadIdx.x] = s;
  __syncthreads();
  for (int m = 128; m > 0; m >>= 1) {
    if (threadIdx.x < m) red[threadIdx.x] += red[threadIdx.x + m];
    __syncthreads();
  }
  if (threadIdx.x == 0) {
    const float mse  = acc2[0] / (float)(rows * DCC);
    const float entm = acc2[1] / (float)rows;
    loss_out[0] = 2.f * mse + (entm - red[0]);   // commit=1, ent_w=1, gamma=1
  }
}

// ---------------------------------------------------------------------------
// Host orchestration
// ---------------------------------------------------------------------------
struct Cur {
  void* const* p;
  int i;
  const float* f() { return (const float*)p[i++]; }
};
struct AttnP { const float *wq,*bq,*wk,*bk,*wv,*bv,*wo,*bo; };
struct EncL  { AttnP sa; const float *g1,*b1,*g2,*b2,*w1,*fb1,*w2,*fb2; };
struct DecL  { AttnP sa; const float *g1,*b1,*g2,*b2,*w1,*fb1,*w2,*fb2; AttnP ca; const float *g3,*b3; };

static AttnP rdAttn(Cur& c) {
  AttnP a;
  a.wq=c.f(); a.bq=c.f(); a.wk=c.f(); a.bk=c.f();
  a.wv=c.f(); a.bv=c.f(); a.wo=c.f(); a.bo=c.f();
  return a;
}
static EncL rdEnc(Cur& c) {
  EncL e;
  e.sa=rdAttn(c);
  e.g1=c.f(); e.b1=c.f(); e.g2=c.f(); e.b2=c.f();
  e.w1=c.f(); e.fb1=c.f(); e.w2=c.f(); e.fb2=c.f();
  return e;
}
static DecL rdDec(Cur& c) {
  DecL d;
  d.sa=rdAttn(c);
  d.g1=c.f(); d.b1=c.f(); d.g2=c.f(); d.b2=c.f();
  d.w1=c.f(); d.fb1=c.f(); d.w2=c.f(); d.fb2=c.f();
  d.ca=rdAttn(c);
  d.g3=c.f(); d.b3=c.f();
  return d;
}

struct Wbuf { float *Qb,*Kb,*Vb,*Ob,*Y,*F; };

static inline void gemmW(const float* A, const float* W, const float* bias, float* C,
                         int M, int N, int K, int act, hipStream_t s)
{
  dim3 g(N / 64, (M + 127) / 128);
  gemm_bf16_wmma<<<g, 256, 0, s>>>(A, W, bias, C, M, N, K, act);
}

static void run_attn(const float* xq, const float* xkv, int Bn, int Lq, int Lk,
                     const AttnP& p, float* out, Wbuf& w, hipStream_t s)
{
  const int Mq = Bn * Lq, Mk = Bn * Lk;
  gemmW(xq,  p.wq, p.bq, w.Qb, Mq, EE, EE, ACT_NONE, s);
  gemmW(xkv, p.wk, p.bk, w.Kb, Mk, EE, EE, ACT_NONE, s);
  gemmW(xkv, p.wv, p.bv, w.Vb, Mk, EE, EE, ACT_NONE, s);
  attn_wmma<<<Bn * NHH, 128, 0, s>>>(w.Qb, w.Kb, w.Vb, w.Ob, Lq, Lk, EE, NHH,
                                     0.17677669529663687f);   // 1/sqrt(32)
  gemmW(w.Ob, p.wo, p.bo, out, Mq, EE, EE, ACT_NONE, s);
}

static void run_enc_layer(float* x, int Bn, int L, const EncL& p, Wbuf& w, hipStream_t s)
{
  const int M = Bn * L;
  run_attn(x, x, Bn, L, L, p.sa, w.Y, w, s);
  add_ln256<<<(M + 7) / 8, 256, 0, s>>>(x, w.Y, p.g1, p.b1, x, M);
  gemmW(x, p.w1, p.fb1, w.F, M, 4 * EE, EE, ACT_GELU, s);
  gemmW(w.F, p.w2, p.fb2, w.Y, M, EE, 4 * EE, ACT_NONE, s);
  add_ln256<<<(M + 7) / 8, 256, 0, s>>>(x, w.Y, p.g2, p.b2, x, M);
}

static void run_dec_layer(float* t, const float* mem, int Bn, int L, int Lm,
                          const DecL& p, Wbuf& w, hipStream_t s)
{
  const int M = Bn * L;
  run_attn(t, t, Bn, L, L, p.sa, w.Y, w, s);
  add_ln256<<<(M + 7) / 8, 256, 0, s>>>(t, w.Y, p.g1, p.b1, t, M);
  run_attn(t, mem, Bn, L, Lm, p.ca, w.Y, w, s);
  add_ln256<<<(M + 7) / 8, 256, 0, s>>>(t, w.Y, p.g3, p.b3, t, M);
  gemmW(t, p.w1, p.fb1, w.F, M, 4 * EE, EE, ACT_GELU, s);
  gemmW(w.F, p.w2, p.fb2, w.Y, M, EE, 4 * EE, ACT_NONE, s);
  add_ln256<<<(M + 7) / 8, 256, 0, s>>>(t, w.Y, p.g2, p.b2, t, M);
}

extern "C" void kernel_launch(void* const* d_in, const int* in_sizes, int n_in,
                              void* d_out, int out_size, void* d_ws, size_t ws_size,
                              hipStream_t stream)
{
  (void)in_sizes; (void)n_in; (void)out_size; (void)ws_size;

  // ---- parameter walk (setup_inputs dict order) ----
  Cur c{d_in, 0};
  const float* x = c.f();
  const float* pw = c.f(); const float* pb = c.f(); const float* pos = c.f();
  EncL enc[NLL]; for (int i = 0; i < NLL; i++) enc[i] = rdEnc(c);
  const float* tokens = c.f();
  DecL tok[NLL]; for (int i = 0; i < NLL; i++) tok[i] = rdDec(c);
  const float* adw1 = c.f(); const float* adb1 = c.f();
  const float* adw2 = c.f(); const float* adb2 = c.f();
  const float* cb = c.f();
  const float* aew = c.f(); const float* aeb = c.f();
  const float* cls = c.f();
  EncL dec[NLL]; for (int i = 0; i < NLL; i++) dec[i] = rdEnc(c);
  const float* decow = c.f(); const float* decob = c.f();
  const float* sws[3]; for (int i = 0; i < 3; i++) sws[i] = c.f();
  const float* sbs[3]; for (int i = 0; i < 3; i++) sbs[i] = c.f();
  const float* sgs[3]; for (int i = 0; i < 3; i++) sgs[i] = c.f();
  const float* slb[3]; for (int i = 0; i < 3; i++) slb[i] = c.f();
  const float* sow = c.f(); const float* sob = c.f();

  // ---- workspace layout ----
  const size_t NT = (size_t)BB * NPP;        // 32768 tokens
  const size_t ND = (size_t)BB * (NPP + 1);  // 33280 tokens (CLS + patches)
  float* Wp = (float*)d_ws;
  size_t off = 0;
  auto alloc = [&](size_t n) { float* r = Wp + off; off += n; return r; };
  float* X    = alloc(NT * EE);
  float* Tt   = alloc(NT * EE);
  float* Dq   = alloc(ND * EE);
  float* Qb   = alloc(ND * EE);
  float* Kb   = alloc(ND * EE);
  float* Vb   = alloc(ND * EE);
  float* Ob   = alloc(ND * EE);
  float* Y    = alloc(ND * EE);
  float* F    = alloc(ND * 4 * EE);
  float* Z    = alloc(NT * DCC);
  float* ZQ   = alloc(NT * DCC);
  float* rowm = alloc(NT);
  float* rowz = alloc(NT);
  float* pmb  = alloc(KK);
  float* acc2 = alloc(8);
  float* S1   = alloc(NT * 32);
  float* S2   = alloc(NT * 32);
  float* Ctmp = alloc(ND * PSS);
  Wbuf wb{Qb, Kb, Vb, Ob, Y, F};

  // ---- output layout: x_hat | loss | patches | idx ----
  const size_t XHAT_N  = (size_t)BB * TT;          // 262144
  const size_t PATCH_N = (size_t)BB * NPP * PSS;   // 262144
  float* out_xhat  = (float*)d_out;
  float* out_loss  = out_xhat + XHAT_N;
  float* out_patch = out_xhat + XHAT_N + 1;
  int*   out_idx   = (int*)(out_xhat + XHAT_N + 1 + PATCH_N);

  const int MT = (int)NT, MD = (int)ND;

  // 1) PatchEncoder: project patches (K=8), add positional embedding
  gemm_small<<<(MT * EE + 255) / 256, 256, 0, stream>>>(x, pw, pb, X, MT, EE, PSS, ACT_NONE);
  add_pos<<<(MT * EE + 255) / 256, 256, 0, stream>>>(X, pos, MT * EE);
  for (int i = 0; i < NLL; i++) run_enc_layer(X, BB, NPP, enc[i], wb, stream);

  // 2) Tokenizer: learned queries cross-attend to encoder memory
  broadcast_tok<<<(MT * EE + 255) / 256, 256, 0, stream>>>(tokens, Tt, MT * EE);
  for (int i = 0; i < NLL; i++) run_dec_layer(Tt, X, BB, NPP, NPP, tok[i], wb, stream);

  // 3) AttributeDecoder MLP -> z (dim 8)
  gemmW(Tt, adw1, adb1, Y, MT, EE, EE, ACT_GELU, stream);
  gemm_small<<<(MT * DCC + 255) / 256, 256, 0, stream>>>(Y, adw2, adb2, Z, MT, DCC, EE, ACT_NONE);

  // 4) EuclCodebook quantize + loss
  zero_f<<<1, 256, 0, stream>>>(acc2, 8);
  zero_f<<<(KK + 255) / 256, 256, 0, stream>>>(pmb, KK);
  cb_pass1<<<(MT + 255) / 256, 256, 0, stream>>>(Z, cb, ZQ, out_idx, rowm, rowz, acc2, MT);
  cb_pass2<<<KK, 256, 0, stream>>>(Z, cb, rowm, rowz, pmb, MT);
  cb_finalize<<<1, 256, 0, stream>>>(acc2, pmb, out_loss, MT);

  // 5) AttributeEncoder -> E, PatchDecoder with CLS token
  gemm_small<<<(MT * EE + 255) / 256, 256, 0, stream>>>(ZQ, aew, aeb, Y, MT, EE, DCC, ACT_NONE);
  build_dseq<<<(MD * EE + 255) / 256, 256, 0, stream>>>(cls, Y, Dq, MD * EE);
  for (int i = 0; i < NLL; i++) run_enc_layer(Dq, BB, NPP + 1, dec[i], wb, stream);
  gemm_small<<<(MD * PSS + 255) / 256, 256, 0, stream>>>(Dq, decow, decob, Ctmp, MD, PSS, EE, ACT_NONE);
  extract_xhat<<<((int)XHAT_N + 255) / 256, 256, 0, stream>>>(Ctmp, out_xhat, (int)XHAT_N);

  // 6) ShapeDecoderMLP: (Linear -> LN -> GELU) x3 -> Linear, dims 8->16->32->16->8
  const int dims[4] = {DCC, 2 * DCC, 4 * DCC, 2 * DCC};
  const float* curp = ZQ;
  for (int i = 0; i < 3; i++) {
    gemm_small<<<(MT * dims[i + 1] + 255) / 256, 256, 0, stream>>>(
        curp, sws[i], sbs[i], S1, MT, dims[i + 1], dims[i], ACT_NONE);
    ln_gelu_small<<<(MT + 255) / 256, 256, 0, stream>>>(S1, sgs[i], slb[i], S2, MT, dims[i + 1]);
    curp = S2;
  }
  gemm_small<<<((int)PATCH_N + 255) / 256, 256, 0, stream>>>(
      curp, sow, sob, out_patch, MT, PSS, 2 * DCC, ACT_NONE);
}
